// SimplifiedCausalHAFE_59554016526893
// MI455X (gfx1250) — matrix operand order repaired and checked by reading, not compiled, not run on hardware
//
#include <hip/hip_runtime.h>

// ---------------------------------------------------------------------------
// Problem constants (match reference)
// ---------------------------------------------------------------------------
#define NN   50000      // nodes
#define EE   600000     // edges
#define AA   4096       // aspect rows
#define NET  4          // edge types
#define LNEPS 1e-5f

typedef _Float16 half16 __attribute__((ext_vector_type(16)));
typedef float    float8 __attribute__((ext_vector_type(8)));

static __device__ __forceinline__ unsigned f2key(float x) {
    unsigned b = __float_as_uint(x);
    return (b & 0x80000000u) ? ~b : (b | 0x80000000u);
}
static __device__ __forceinline__ float key2f(unsigned k) {
    return __uint_as_float((k & 0x80000000u) ? (k ^ 0x80000000u) : ~k);
}

// ---------------------------------------------------------------------------
// WMMA GEMM:  C[M,Nc] = act( A[M,K] @ W[K,Nc] + bias )
//   - A rows optionally gathered through rowIdx
//   - f32 inputs converted to f16 in LDS, v_wmma_f32_16x16x32_f16 accumulate
//   - block tile 128x64, 8 waves (4x2), each wave 32x32 (2x2 WMMA tiles)
//   - B staged TRANSPOSED in LDS so both A and B fragment reads are
//     two contiguous 16B runs -> two aligned ds_load_b128 per fragment (no scalar
//     column reads / v_mov_b16 packing on the WMMA critical path)
//   - requires K % 32 == 0, Nc % 64 == 0
// ---------------------------------------------------------------------------
__global__ __launch_bounds__(256)
void k_gemm_wmma(const float* __restrict__ A, const float* __restrict__ W,
                 const float* __restrict__ bias, float* __restrict__ C,
                 const int* __restrict__ rowIdx,
                 int M, int K, int Nc, int act)
{
    __shared__ _Float16 sA [128][40];  // [row][k]   128x32 (+pad)
    __shared__ _Float16 sBt[64][40];   // [col][k]   64x32 transposed (+pad)

    const int tid  = threadIdx.x;
    const int lane = tid & 31;
    const int wid  = tid >> 5;          // wave32: 8 waves per block
    const int wm   = (wid >> 1) * 32;   // wave M offset in block tile
    const int wn   = (wid & 1)  * 32;   // wave N offset in block tile
    const int m0   = blockIdx.y * 128;
    const int n0   = blockIdx.x * 64;
    const int lr   = lane & 15;
    const int kA   = (lane < 16) ? 0 : 8;    // A frag K split (ISA 7.12.2)
    const int kB   = (lane < 16) ? 0 : 16;   // unused; B frag uses kA too

    float8 acc[2][2];
#pragma unroll
    for (int i = 0; i < 2; ++i)
#pragma unroll
        for (int j = 0; j < 2; ++j)
#pragma unroll
            for (int r = 0; r < 8; ++r) acc[i][j][r] = 0.0f;

    for (int k0 = 0; k0 < K; k0 += 32) {
        // ---- stage A tile (128x32 f32 -> f16), 16 elems/thread ----
        {
            const int kk = (tid & 7) * 4;
            const int rb = tid >> 3;
#pragma unroll
            for (int r = 0; r < 4; ++r) {
                const int row = rb + 32 * r;
                const int gm  = m0 + row;
                float4 v = make_float4(0.f, 0.f, 0.f, 0.f);
                if (gm < M) {
                    const int gr = rowIdx ? rowIdx[gm] : gm;
                    const float* p = A + (size_t)gr * K + k0 + kk;
                    v = *(const float4*)p;
                    if (k0 + 32 < K) __builtin_prefetch(p + 32, 0, 3); // global_prefetch_b8 (near)
                }
                sA[row][kk + 0] = (_Float16)v.x;
                sA[row][kk + 1] = (_Float16)v.y;
                sA[row][kk + 2] = (_Float16)v.z;
                sA[row][kk + 3] = (_Float16)v.w;
            }
        }
        // ---- stage B tile (32x64 f32 -> f16), transposed into sBt[col][k] ----
        {
            const int col = (tid & 15) * 4;
            const int rb  = tid >> 4;
#pragma unroll
            for (int r = 0; r < 2; ++r) {
                const int row = rb + 16 * r;   // k within tile
                const float* p = W + (size_t)(k0 + row) * Nc + n0 + col;
                float4 v = *(const float4*)p;
                sBt[col + 0][row] = (_Float16)v.x;
                sBt[col + 1][row] = (_Float16)v.y;
                sBt[col + 2][row] = (_Float16)v.z;
                sBt[col + 3][row] = (_Float16)v.w;
            }
        }
        __syncthreads();

        // ---- build fragments (two contiguous 16B reads each) + 4 WMMAs ----
        half16 af[2], bf[2];
#pragma unroll
        for (int i = 0; i < 2; ++i) {
            const int row = wm + i * 16 + lr;
#pragma unroll
            for (int e = 0; e < 8; ++e) {
                af[i][e]     = sA[row][kA + e];       // K = kA .. kA+7
                af[i][8 + e] = sA[row][kA + 16 + e];  // K = kA+16 .. kA+23
            }
        }
#pragma unroll
        for (int j = 0; j < 2; ++j) {
            const int col = wn + j * 16 + lr;
#pragma unroll
            for (int e = 0; e < 16; ++e) bf[j][e] = sBt[col][(e < 8) ? (kA + e) : (kA + 8 + e)];  // K = kA..kA+7, kA+16..kA+23
        }
#pragma unroll
        for (int i = 0; i < 2; ++i)
#pragma unroll
            for (int j = 0; j < 2; ++j)
                acc[i][j] = __builtin_amdgcn_wmma_f32_16x16x32_f16(
                    false, af[i], false, bf[j], (short)0, acc[i][j], false, false);
        __syncthreads();
    }

    // ---- epilogue: bias + activation + store (C/D layout per ISA 7.12.2) ----
    const int ro = (lane < 16) ? 0 : 8;
#pragma unroll
    for (int i = 0; i < 2; ++i) {
#pragma unroll
        for (int j = 0; j < 2; ++j) {
            const int gc = n0 + wn + j * 16 + lr;
            const float bv = bias ? bias[gc] : 0.f;
#pragma unroll
            for (int r = 0; r < 8; ++r) {
                const int gr = m0 + wm + i * 16 + ro + r;
                if (gr < M) {
                    float v = acc[i][j][r] + bv;
                    if (act) v = fmaxf(v, 0.f);
                    C[(size_t)gr * Nc + gc] = v;
                }
            }
        }
    }
}

// ---------------------------------------------------------------------------
// h += type_emb[node_type] + position * pos_w + pos_b     (D = 256)
// ---------------------------------------------------------------------------
__global__ void k_mix(float* __restrict__ h, const int* __restrict__ nt,
                      const float* __restrict__ pos, const float* __restrict__ te,
                      const float* __restrict__ pw, const float* __restrict__ pb,
                      int N, int D)
{
    long long idx = (long long)blockIdx.x * blockDim.x + threadIdx.x;
    if (idx >= (long long)N * D) return;
    const int n = (int)(idx / D), c = (int)(idx % D);
    h[idx] += te[nt[n] * D + c] + pos[n] * pw[c] + pb[c];
}

// ---------------------------------------------------------------------------
// per-node attention logits: als/ald[n,h] = <xh[n,h,:], att_src/dst[h,:]>
// ---------------------------------------------------------------------------
__global__ void k_node_att(const float* __restrict__ xh, const float* __restrict__ as,
                           const float* __restrict__ ad, float* __restrict__ als,
                           float* __restrict__ ald, int N, int H, int C)
{
    int idx = blockIdx.x * blockDim.x + threadIdx.x;
    if (idx >= N * H) return;
    const int h = idx % H, n = idx / H;
    const float* x = xh + (size_t)n * H * C + h * C;
    float sa = 0.f, sd = 0.f;
    for (int c = 0; c < C; ++c) { float v = x[c]; sa += v * as[h * C + c]; sd += v * ad[h * C + c]; }
    als[idx] = sa; ald[idx] = sd;
}

// edge-type histogram (for the self-loop mean edge attribute)
__global__ void k_hist(const int* __restrict__ et, int E, int* __restrict__ cnt)
{
    int i = blockIdx.x * blockDim.x + threadIdx.x;
    if (i < E) atomicAdd(&cnt[et[i]], 1);
}

// al_e table for the NET edge types + 1 mean (self-loop) row
__global__ void k_tab(const float* __restrict__ emb, const float* __restrict__ lw,
                      const float* __restrict__ ae, const int* __restrict__ cnt,
                      int E, float* __restrict__ tab, int H, int C)
{
    const int id = threadIdx.x;
    if (id >= (NET + 1) * H) return;
    const int t = id / H, h = id % H;
    float ea[NET];
    if (t < NET) {
        for (int k = 0; k < NET; ++k) ea[k] = emb[t * NET + k];
    } else {
        for (int k = 0; k < NET; ++k) {
            float s = 0.f;
            for (int tt = 0; tt < NET; ++tt) s += (float)cnt[tt] * emb[tt * NET + k];
            ea[k] = s / (float)E;
        }
    }
    float a = 0.f;
    for (int c = 0; c < C; ++c) {
        float eh = 0.f;
        for (int k = 0; k < NET; ++k) eh += ea[k] * lw[k * H * C + h * C + c];
        a += eh * ae[h * C + c];
    }
    tab[id] = a;
}

// leaky-relu logits + running segment max (monotone uint keys, atomicMax)
__global__ void k_edge_logits(const int* __restrict__ src, const int* __restrict__ dst,
                              const int* __restrict__ et, const float* __restrict__ als,
                              const float* __restrict__ ald, const float* __restrict__ tab,
                              float* __restrict__ elog, unsigned* __restrict__ dmax,
                              int E, int N, int H)
{
    long long idx = (long long)blockIdx.x * blockDim.x + threadIdx.x;
    if (idx >= (long long)(E + N) * H) return;
    const int h = (int)(idx % H);
    const int e = (int)(idx / H);
    int s, d, t;
    if (e < E) { s = src[e]; d = dst[e]; t = et[e]; }
    else       { s = d = e - E; t = NET; }
    float x = als[s * H + h] + ald[d * H + h] + tab[t * H + h];
    x = (x > 0.f) ? x : 0.2f * x;
    elog[idx] = x;
    atomicMax(&dmax[d * H + h], f2key(x));
}

// exp(logit - max) in place + segment sum
__global__ void k_edge_exp(const int* __restrict__ dst, float* __restrict__ elog,
                           const unsigned* __restrict__ dmax, float* __restrict__ dsum,
                           int E, int N, int H)
{
    long long idx = (long long)blockIdx.x * blockDim.x + threadIdx.x;
    if (idx >= (long long)(E + N) * H) return;
    const int h = (int)(idx % H);
    const int e = (int)(idx / H);
    const int d = (e < E) ? dst[e] : e - E;
    const float m  = key2f(dmax[d * H + h]);
    const float ev = __expf(elog[idx] - m);
    elog[idx] = ev;
    atomicAdd(&dsum[d * H + h], ev);
}

// weighted message scatter: agg[d,h,:] += alpha * xh[s,h,:]   (4 ch / thread)
__global__ void k_scatter(const int* __restrict__ src, const int* __restrict__ dst,
                          const float* __restrict__ xh, const float* __restrict__ elog,
                          const float* __restrict__ dsum, float* __restrict__ agg,
                          int E, int N, int H, int C)
{
    const int c4n = C >> 2;
    long long idx = (long long)blockIdx.x * blockDim.x + threadIdx.x;
    if (idx >= (long long)(E + N) * H * c4n) return;
    const int c4 = (int)(idx % c4n);
    long long r  = idx / c4n;
    const int h  = (int)(r % H);
    const int e  = (int)(r / H);
    int s, d;
    if (e < E) { s = src[e]; d = dst[e]; } else { s = d = e - E; }
    const float w = elog[(size_t)e * H + h] / (dsum[d * H + h] + 1e-16f);
    const float4 xv = *(const float4*)(xh + (size_t)s * H * C + h * C + c4 * 4);
    float* o = agg + (size_t)d * H * C + h * C + c4 * 4;
    atomicAdd(o + 0, xv.x * w);
    atomicAdd(o + 1, xv.y * w);
    atomicAdd(o + 2, xv.z * w);
    atomicAdd(o + 3, xv.w * w);
}

// out = LN(agg + bias + res) * g + b  (optional relu), one block per row
__global__ void k_ln(const float* __restrict__ agg, const float* __restrict__ res,
                     const float* __restrict__ bias, const float* __restrict__ g,
                     const float* __restrict__ b, float* __restrict__ out,
                     int D, int act)
{
    const int n = blockIdx.x;
    __shared__ float r1[128], r2[128];
    float s1 = 0.f, s2 = 0.f;
    for (int c = threadIdx.x; c < D; c += 128) {
        float v = agg[(size_t)n * D + c] + bias[c] + res[(size_t)n * D + c];
        s1 += v; s2 += v * v;
    }
    r1[threadIdx.x] = s1; r2[threadIdx.x] = s2;
    __syncthreads();
    for (int o = 64; o > 0; o >>= 1) {
        if (threadIdx.x < o) { r1[threadIdx.x] += r1[threadIdx.x + o]; r2[threadIdx.x] += r2[threadIdx.x + o]; }
        __syncthreads();
    }
    const float mean = r1[0] / (float)D;
    const float var  = r2[0] / (float)D - mean * mean;
    const float inv  = rsqrtf(var + LNEPS);
    for (int c = threadIdx.x; c < D; c += 128) {
        float v = agg[(size_t)n * D + c] + bias[c] + res[(size_t)n * D + c];
        float o = (v - mean) * inv * g[c] + b[c];
        if (act) o = fmaxf(o, 0.f);
        out[(size_t)n * D + c] = o;
    }
}

// jin[a] = concat(g2out[aspect[a]] (128), zs[a] (64))
__global__ void k_concat(const float* __restrict__ g2o, const float* __restrict__ zs,
                         const int* __restrict__ asp, float* __restrict__ jin, int A)
{
    int idx = blockIdx.x * blockDim.x + threadIdx.x;
    if (idx >= A * 192) return;
    const int a = idx / 192, c = idx % 192;
    jin[idx] = (c < 128) ? g2o[(size_t)asp[a] * 128 + c] : zs[(size_t)a * 64 + (c - 128)];
}

// final 256 -> 3 output layer
__global__ void k_cls3(const float* __restrict__ x, const float* __restrict__ w,
                       const float* __restrict__ b, float* __restrict__ out, int A)
{
    int idx = blockIdx.x * blockDim.x + threadIdx.x;
    if (idx >= A * 3) return;
    const int a = idx / 3, j = idx % 3;
    float s = b[j];
    for (int k = 0; k < 256; ++k) s += x[(size_t)a * 256 + k] * w[k * 3 + j];
    out[idx] = s;
}

// ---------------------------------------------------------------------------
// Host orchestration
// ---------------------------------------------------------------------------
enum {
    IN_FEAT = 0, IN_EIDX, IN_ASP, IN_ETYPE, IN_NTYPE, IN_POS,
    P_FE_W1, P_FE_B1, P_FE_W2, P_FE_B2, P_FE_POS_W, P_FE_POS_B, P_FE_TYPE_EMB,
    P_FE_OUT_W, P_FE_OUT_B,
    P_DIB_WC, P_DIB_BC, P_DIB_WS, P_DIB_BS,
    P_G1_LIN_W, P_G1_ATT_SRC, P_G1_ATT_DST, P_G1_LIN_EDGE_W, P_G1_ATT_EDGE,
    P_G1_BIAS, P_G1_EDGE_EMB, P_G1_RES_W, P_G1_RES_B, P_G1_LN_G, P_G1_LN_B,
    P_G2_LIN_W, P_G2_ATT_SRC, P_G2_ATT_DST, P_G2_LIN_EDGE_W, P_G2_ATT_EDGE,
    P_G2_BIAS, P_G2_EDGE_EMB, P_G2_RES_W, P_G2_RES_B, P_G2_LN_G, P_G2_LN_B,
    P_CLS_W1, P_CLS_B1, P_CLS_W2, P_CLS_B2, P_CLS_W3, P_CLS_B3
};

static inline long long cdivll(long long a, long long b) { return (a + b - 1) / b; }

extern "C" void kernel_launch(void* const* d_in, const int* in_sizes, int n_in,
                              void* d_out, int out_size, void* d_ws, size_t ws_size,
                              hipStream_t stream)
{
    (void)in_sizes; (void)n_in; (void)out_size; (void)ws_size;

    const float* feat  = (const float*)d_in[IN_FEAT];
    const int*   src   = (const int*)d_in[IN_EIDX];
    const int*   dst   = src + EE;
    const int*   asp   = (const int*)d_in[IN_ASP];
    const int*   etype = (const int*)d_in[IN_ETYPE];
    const int*   ntype = (const int*)d_in[IN_NTYPE];
    const float* pos   = (const float*)d_in[IN_POS];
    #define PF(i) ((const float*)d_in[(i)])

    // ---- workspace bump allocator (floats) ----
    float* wsf = (float*)d_ws;
    size_t off = 0;
    auto alloc = [&](size_t nf) { float* p = wsf + off; off += nf; return p; };
    float* h1   = alloc((size_t)NN * 256);   // fe1 out, then enh
    float* h2   = alloc((size_t)NN * 256);
    float* zc   = alloc((size_t)NN * 128);
    float* zs   = alloc((size_t)AA * 64);
    float* xh   = alloc((size_t)NN * 512);   // layer1 H*C=512; layer2 reuses 128
    float* res  = alloc((size_t)NN * 512);
    float* agg  = alloc((size_t)NN * 512);
    float* gact = alloc((size_t)NN * 512);
    float* g2o  = alloc((size_t)NN * 128);
    float* als  = alloc((size_t)NN * 4);
    float* ald  = alloc((size_t)NN * 4);
    float* elog = alloc((size_t)(EE + NN) * 4);
    unsigned* dmax = (unsigned*)alloc((size_t)NN * 4);
    float* dsum = alloc((size_t)NN * 4);
    float* tab  = alloc(64);
    int*   cnt  = (int*)alloc(16);
    float* jin  = alloc((size_t)AA * 192);
    float* c1   = alloc((size_t)AA * 256);
    float* c2   = alloc((size_t)AA * 256);

    auto gemm = [&](const float* Am, const float* Wm, const float* bm, float* Cm,
                    const int* ridx, int M, int K, int Nc, int act) {
        dim3 g((unsigned)(Nc / 64), (unsigned)cdivll(M, 128));
        k_gemm_wmma<<<g, dim3(256), 0, stream>>>(Am, Wm, bm, Cm, ridx, M, K, Nc, act);
    };
    auto run_gat = [&](const float* xhl, const float* resl, const float* embp,
                       const float* lewp, const float* aep, const float* asp_w,
                       const float* adp, const float* biasp, const float* lng,
                       const float* lnb, float* outp, int H, int C, int act) {
        const int HC = H * C;
        k_node_att<<<(unsigned)cdivll((long long)NN * H, 256), 256, 0, stream>>>(
            xhl, asp_w, adp, als, ald, NN, H, C);
        k_tab<<<1, 32, 0, stream>>>(embp, lewp, aep, cnt, EE, tab, H, C);
        hipMemsetAsync(dmax, 0, (size_t)NN * H * sizeof(unsigned), stream);
        hipMemsetAsync(dsum, 0, (size_t)NN * H * sizeof(float), stream);
        hipMemsetAsync(agg, 0, (size_t)NN * HC * sizeof(float), stream);
        const long long eh = (long long)(EE + NN) * H;
        k_edge_logits<<<(unsigned)cdivll(eh, 256), 256, 0, stream>>>(
            src, dst, etype, als, ald, tab, elog, dmax, EE, NN, H);
        k_edge_exp<<<(unsigned)cdivll(eh, 256), 256, 0, stream>>>(
            dst, elog, dmax, dsum, EE, NN, H);
        k_scatter<<<(unsigned)cdivll(eh * (C / 4), 256), 256, 0, stream>>>(
            src, dst, xhl, elog, dsum, agg, EE, NN, H, C);
        k_ln<<<NN, 128, 0, stream>>>(agg, resl, biasp, lng, lnb, outp, HC, act);
    };

    // ---- feature extractor ----
    gemm(feat, PF(P_FE_W1), PF(P_FE_B1), h1, nullptr, NN, 768, 256, 1);
    gemm(h1,   PF(P_FE_W2), PF(P_FE_B2), h2, nullptr, NN, 256, 256, 0);
    k_mix<<<(unsigned)cdivll((long long)NN * 256, 256), 256, 0, stream>>>(
        h2, ntype, pos, PF(P_FE_TYPE_EMB), PF(P_FE_POS_W), PF(P_FE_POS_B), NN, 256);
    gemm(h2, PF(P_FE_OUT_W), PF(P_FE_OUT_B), h1, nullptr, NN, 256, 256, 0);   // enh -> h1

    // ---- DIB split ----
    gemm(h1, PF(P_DIB_WC), PF(P_DIB_BC), zc, nullptr, NN, 256, 128, 0);
    gemm(h1, PF(P_DIB_WS), PF(P_DIB_BS), zs, asp,     AA, 256, 64,  0);

    // ---- edge-type histogram (shared by both GAT layers) ----
    hipMemsetAsync(cnt, 0, NET * sizeof(int), stream);
    k_hist<<<(unsigned)cdivll(EE, 256), 256, 0, stream>>>(etype, EE, cnt);

    // ---- GAT layer 1: heads=4, C=128 ----
    gemm(zc, PF(P_G1_LIN_W), nullptr,        xh,  nullptr, NN, 128, 512, 0);
    gemm(zc, PF(P_G1_RES_W), PF(P_G1_RES_B), res, nullptr, NN, 128, 512, 0);
    run_gat(xh, res, PF(P_G1_EDGE_EMB), PF(P_G1_LIN_EDGE_W), PF(P_G1_ATT_EDGE),
            PF(P_G1_ATT_SRC), PF(P_G1_ATT_DST), PF(P_G1_BIAS),
            PF(P_G1_LN_G), PF(P_G1_LN_B), gact, 4, 128, /*relu=*/1);

    // ---- GAT layer 2: heads=1, C=128 ----
    gemm(gact, PF(P_G2_LIN_W), nullptr,        xh,  nullptr, NN, 512, 128, 0);
    gemm(gact, PF(P_G2_RES_W), PF(P_G2_RES_B), res, nullptr, NN, 512, 128, 0);
    run_gat(xh, res, PF(P_G2_EDGE_EMB), PF(P_G2_LIN_EDGE_W), PF(P_G2_ATT_EDGE),
            PF(P_G2_ATT_SRC), PF(P_G2_ATT_DST), PF(P_G2_BIAS),
            PF(P_G2_LN_G), PF(P_G2_LN_B), g2o, 1, 128, /*relu=*/0);

    // ---- joint classifier on aspect rows ----
    k_concat<<<(unsigned)cdivll((long long)AA * 192, 256), 256, 0, stream>>>(
        g2o, zs, asp, jin, AA);
    gemm(jin, PF(P_CLS_W1), PF(P_CLS_B1), c1, nullptr, AA, 192, 256, 1);
    gemm(c1,  PF(P_CLS_W2), PF(P_CLS_B2), c2, nullptr, AA, 256, 256, 1);
    k_cls3<<<(unsigned)cdivll((long long)AA * 3, 256), 256, 0, stream>>>(
        c2, PF(P_CLS_W3), PF(P_CLS_B3), (float*)d_out, AA);
}
